// attentiond_35390530519835
// MI455X (gfx1250) — compile-verified, matrix operand order checked by reading, not run on hardware
//
#include <hip/hip_runtime.h>

// ---------------------------------------------------------------------------
// CDNA5 (gfx1250) fused attentiond forward, bf16 WMMA + f32 accumulate.
//   k_cvt_t : W_kv, W_out f32 -> bf16 TRANSPOSED (BT[N][K]) one-time
//   k_ln    : h = LayerNorm(x) -> bf16
//   k_gemm  : kv = h @ W_kv -> qk [b,h,n,64] (row-major) + vT [b,h,64,n]
//   k_stats : per-row softmax max/sum of S = qk qk^T (WMMA, streamed)
//   k_attn  : recompute S, p = exp(s-m)/l, W_x head-mix via LDS,
//             o1 += Pmix @ v (WMMA, B from vT) -> o1 bf16 [b,n,1024]
//   k_gemm  : out = o1 @ W_out -> f32
//
// Fragment I/O exploits the ISA 7.12.2 layout: a lane's 16 halves of a
// 16x32 bf16 A/B fragment are two contiguous 8-half runs at K = 8*lhi and
// K = 16 + 8*lhi of one row -> two 16-byte loads per fragment. GEMM tile
// staging uses GLOBAL_LOAD_ASYNC_TO_LDS_B128 (ASYNCcnt) when the toolchain
// exposes it, else b128 VGPR staging.
// ---------------------------------------------------------------------------

#define NCTX 2048
#define NCTX_SHIFT 11
#define NCTX_MASK 2047

#if __has_builtin(__builtin_amdgcn_global_load_async_to_lds_b128) && \
    __has_builtin(__builtin_amdgcn_s_wait_asynccnt)
#define USE_ASYNC_LDS 1
#else
#define USE_ASYNC_LDS 0
#endif

typedef __attribute__((ext_vector_type(16))) __bf16 v16bf;
typedef __attribute__((ext_vector_type(8)))  float  v8f;
typedef __attribute__((ext_vector_type(4)))  int    v4i;

union V16 { v16bf v; __bf16 e[16]; uint4 q[2]; };
union V8  { v8f   v; float  e[8];  };

static __device__ inline unsigned int f2bf_bits(float f) {
    unsigned int u = __float_as_uint(f);
    return (u + 0x7FFFu + ((u >> 16) & 1u)) >> 16;   // round-to-nearest-even
}
static __device__ inline __bf16 f2bf(float f) {
    union { unsigned short u; __bf16 b; } c;
    c.u = (unsigned short)f2bf_bits(f);
    return c.b;
}
static __device__ inline uint4 pack8(const float* f) {
    uint4 o;
    o.x = f2bf_bits(f[0]) | (f2bf_bits(f[1]) << 16);
    o.y = f2bf_bits(f[2]) | (f2bf_bits(f[3]) << 16);
    o.z = f2bf_bits(f[4]) | (f2bf_bits(f[5]) << 16);
    o.w = f2bf_bits(f[6]) | (f2bf_bits(f[7]) << 16);
    return o;
}

// one 16B tile-staging copy: global -> LDS
static __device__ inline void stage_b128(const __bf16* g, __bf16* l) {
#if USE_ASYNC_LDS
    __builtin_amdgcn_global_load_async_to_lds_b128(
        (__attribute__((address_space(1))) v4i*)g,
        (__attribute__((address_space(3))) v4i*)l, 0, 0);
#else
    *(uint4*)l = *(const uint4*)g;
#endif
}
static __device__ inline void wait_stage() {
#if USE_ASYNC_LDS
    __builtin_amdgcn_s_wait_asynccnt(0);
#endif
}

// Load one 16x32 bf16 A/B fragment from a row-major source.
// rowptr = &src[row * stride + k0]; must be 16B aligned.
static __device__ inline v16bf load_frag(const __bf16* rowptr, int lhi) {
    V16 u;
    u.q[0] = *(const uint4*)(rowptr + 8 * lhi);
    u.q[1] = *(const uint4*)(rowptr + 16 + 8 * lhi);
    return u.v;
}

static __device__ inline v8f wmma_bf16(v16bf a, v16bf b, v8f c) {
    return __builtin_amdgcn_wmma_f32_16x16x32_bf16(
        false, a, false, b, (short)0, c, false, false);
}

// --------------------- f32 -> bf16 with transpose --------------------------
// in: [K][N] f32 row-major; out: [N][K] bf16. grid (N/256, K), block 256.
__global__ void k_cvt_t(const float* __restrict__ in, __bf16* __restrict__ out,
                        int N, int K) {
    int n = blockIdx.x * 256 + threadIdx.x;
    int k = blockIdx.y;
    out[(size_t)n * K + k] = f2bf(in[(size_t)k * N + n]);
}

// ------------------------------ LayerNorm ----------------------------------
// one block (256 threads) per row of 1024
__global__ void __launch_bounds__(256) k_ln(
        const float* __restrict__ x, const float* __restrict__ g,
        const float* __restrict__ bta, __bf16* __restrict__ h) {
    const int dims = 1024;
    const int row  = blockIdx.x;
    const int tid = threadIdx.x, lane = tid & 31, wid = tid >> 5;

    float4 v4 = *(const float4*)(x + (size_t)row * dims + tid * 4);
    float s  = v4.x + v4.y + v4.z + v4.w;
    float sq = v4.x * v4.x + v4.y * v4.y + v4.z * v4.z + v4.w * v4.w;
#pragma unroll
    for (int o = 16; o > 0; o >>= 1) {
        s  += __shfl_down(s,  o, 32);
        sq += __shfl_down(sq, o, 32);
    }
    __shared__ float red_s[8], red_q[8];
    __shared__ float mu_s, rstd_s;
    if (lane == 0) { red_s[wid] = s; red_q[wid] = sq; }
    __syncthreads();
    if (tid == 0) {
        float ts = 0.f, tq = 0.f;
        for (int w = 0; w < 8; ++w) { ts += red_s[w]; tq += red_q[w]; }
        float mu  = ts / dims;
        mu_s   = mu;
        rstd_s = rsqrtf(tq / dims - mu * mu + 1e-5f);
    }
    __syncthreads();
    const float mu = mu_s, rstd = rstd_s;
    float4 g4 = *(const float4*)(g + tid * 4);
    float4 b4 = *(const float4*)(bta + tid * 4);
    float o[4];
    o[0] = (v4.x - mu) * rstd * g4.x + b4.x;
    o[1] = (v4.y - mu) * rstd * g4.y + b4.y;
    o[2] = (v4.z - mu) * rstd * g4.z + b4.z;
    o[3] = (v4.w - mu) * rstd * g4.w + b4.w;
    uint2 p;
    p.x = f2bf_bits(o[0]) | (f2bf_bits(o[1]) << 16);
    p.y = f2bf_bits(o[2]) | (f2bf_bits(o[3]) << 16);
    *(uint2*)(h + (size_t)row * dims + tid * 4) = p;
}

// ------------------------------ WMMA GEMM ----------------------------------
// C[M,N] = A[M,K] @ B[K,N] with B given TRANSPOSED: BT[N][K] bf16.
// 256 thr = 8 waves (4Mx2N), block tile 64x64, wave tile 16x32, K step 32,
// double-buffered LDS (one barrier per K step), async-to-LDS staging.
// out_mode 0: f32 row-major -> c_f32
// out_mode 2: kv split: col<1024 -> qk[b,h,n,64]; else vT[b,h,64,n] (packed)
__global__ void __launch_bounds__(256) k_gemm(
        const __bf16* __restrict__ A, const __bf16* __restrict__ BT,
        int M, int N, int K,
        float* __restrict__ c_f32,
        __bf16* __restrict__ qk_out, __bf16* __restrict__ vT_out,
        int out_mode) {
    __shared__ __align__(16) __bf16 lds_a[2][64 * 32];   // [m][k]
    __shared__ __align__(16) __bf16 lds_b[2][64 * 32];   // [n][k]

    const int tid  = threadIdx.x;
    const int lane = tid & 31, w = tid >> 5;
    const int lhi  = lane >> 4, mcol = lane & 15;
    const int wm = w & 3, wn = w >> 2;                   // 4 x 2 wave grid
    const int bm = blockIdx.y * 64, bn = blockIdx.x * 64;

    const int sr = tid >> 2, sc = (tid & 3) * 8;         // staging coords
    const __bf16* gA = A  + (size_t)(bm + sr) * K + sc;
    const __bf16* gB = BT + (size_t)(bn + sr) * K + sc;

    V8 c0, c1;
#pragma unroll
    for (int i = 0; i < 8; ++i) { c0.e[i] = 0.f; c1.e[i] = 0.f; }

    const int KT = K >> 5;
    // prologue: stage tile 0 into buffer 0
    stage_b128(gA, &lds_a[0][sr * 32 + sc]);
    stage_b128(gB, &lds_b[0][sr * 32 + sc]);
    wait_stage();
    __syncthreads();

    for (int kt = 0; kt < KT; ++kt) {
        const int cur = kt & 1;
        if (kt + 1 < KT) {                               // stage next tile
            stage_b128(gA + (size_t)(kt + 1) * 32, &lds_a[cur ^ 1][sr * 32 + sc]);
            stage_b128(gB + (size_t)(kt + 1) * 32, &lds_b[cur ^ 1][sr * 32 + sc]);
        }
        v16bf af = load_frag(&lds_a[cur][(wm * 16 + mcol) * 32], lhi);
        v16bf b0 = load_frag(&lds_b[cur][(wn * 32 + mcol) * 32], lhi);
        v16bf b1 = load_frag(&lds_b[cur][(wn * 32 + 16 + mcol) * 32], lhi);
        c0.v = wmma_bf16(af, b0, c0.v);
        c1.v = wmma_bf16(af, b1, c1.v);
        wait_stage();
        __syncthreads();
    }

    const int row0 = bm + wm * 16 + 8 * lhi;             // 8 consecutive rows
    const int b    = row0 >> NCTX_SHIFT;
    const int n0   = row0 & NCTX_MASK;
#pragma unroll
    for (int ct = 0; ct < 2; ++ct) {
        V8& c = ct ? c1 : c0;
        int col = bn + wn * 32 + ct * 16 + mcol;
        if (out_mode == 0) {
#pragma unroll
            for (int r = 0; r < 8; ++r)
                c_f32[(size_t)(row0 + r) * N + col] = c.e[r];
        } else if (col < 1024) {                         // qk: [b,h,n,64]
            int hd = col >> 6, d = col & 63;
            __bf16* dst = qk_out + (((size_t)(b * 16 + hd)) * NCTX) * 64 + d;
#pragma unroll
            for (int r = 0; r < 8; ++r)
                dst[(size_t)(n0 + r) * 64] = f2bf(c.e[r]);
        } else {                                         // vT: [b,h,64,n] packed
            int cc = col - 1024, hd = cc >> 6, d = cc & 63;
            *(uint4*)(vT_out + (((size_t)(b * 16 + hd)) * 64 + d) * NCTX + n0)
                = pack8(c.e);
        }
    }
}

// -------------------------- softmax stats (pass 1) -------------------------
// grid (NCTX/16, b*16). block 256 = 8 waves; wave w covers a k-stripe of 256.
__global__ void __launch_bounds__(256) k_stats(
        const __bf16* __restrict__ qk,
        float* __restrict__ m_out, float* __restrict__ l_out) {
    const int bh = blockIdx.y;
    const int q0 = blockIdx.x * 16;
    const __bf16* qh = qk + (size_t)bh * NCTX * 64;

    const int tid = threadIdx.x, lane = tid & 31, w = tid >> 5;
    const int lhi = lane >> 4, mcol = lane & 15;

    v16bf a0 = load_frag(qh + (size_t)(q0 + mcol) * 64, lhi);
    v16bf a1 = load_frag(qh + (size_t)(q0 + mcol) * 64 + 32, lhi);

    float mrun[8], lrun[8];
#pragma unroll
    for (int r = 0; r < 8; ++r) { mrun[r] = -3.0e38f; lrun[r] = 0.f; }

    const int stripe = NCTX >> 3;
#pragma unroll 2
    for (int kt = w * stripe; kt < (w + 1) * stripe; kt += 16) {
        if (kt + 16 < (w + 1) * stripe)                  // next tile's lines
            __builtin_prefetch(qh + (size_t)(kt + 16 + mcol) * 64, 0, 3);
        v16bf b0 = load_frag(qh + (size_t)(kt + mcol) * 64, lhi);
        v16bf b1 = load_frag(qh + (size_t)(kt + mcol) * 64 + 32, lhi);
        V8 s;
#pragma unroll
        for (int i = 0; i < 8; ++i) s.e[i] = 0.f;
        s.v = wmma_bf16(a0, b0, s.v);
        s.v = wmma_bf16(a1, b1, s.v);
#pragma unroll
        for (int r = 0; r < 8; ++r) {
            float mn = fmaxf(mrun[r], s.e[r]);
            lrun[r] = lrun[r] * __expf(mrun[r] - mn) + __expf(s.e[r] - mn);
            mrun[r] = mn;
        }
    }
#pragma unroll
    for (int o = 1; o < 16; o <<= 1) {                   // merge 16 lanes / half
#pragma unroll
        for (int r = 0; r < 8; ++r) {
            float mo = __shfl_xor(mrun[r], o, 32);
            float lo = __shfl_xor(lrun[r], o, 32);
            float mn = fmaxf(mrun[r], mo);
            lrun[r] = lrun[r] * __expf(mrun[r] - mn) + lo * __expf(mo - mn);
            mrun[r] = mn;
        }
    }
    __shared__ float sm[8][16], sl[8][16];
    if (mcol == 0) {
#pragma unroll
        for (int r = 0; r < 8; ++r) {
            sm[w][r + 8 * lhi] = mrun[r];
            sl[w][r + 8 * lhi] = lrun[r];
        }
    }
    __syncthreads();
    if (tid < 16) {
        float m = -3.0e38f, l = 0.f;
        for (int ww = 0; ww < 8; ++ww) {
            float mn = fmaxf(m, sm[ww][tid]);
            l = l * __expf(m - mn) + sl[ww][tid] * __expf(sm[ww][tid] - mn);
            m = mn;
        }
        m_out[(size_t)bh * NCTX + q0 + tid] = m;
        l_out[(size_t)bh * NCTX + q0 + tid] = l;
    }
}

// ---------------------- attention + head-mix (pass 2) ----------------------
// grid (NCTX/16, b). block 512 = 16 waves, wave h = head h.
#define PBS 36   // padded j-stride (multiple of 4 for float4, !=0 mod 64)
__global__ void __launch_bounds__(512) k_attn(
        const __bf16* __restrict__ qk, const __bf16* __restrict__ vT,
        const float* __restrict__ mstat, const float* __restrict__ lstat,
        const float* __restrict__ Wx, __bf16* __restrict__ o1) {
    __shared__ float wx_s[256];
    __shared__ __align__(16) float pbuf[16 * 16 * PBS]; // [head][row i][k_local]

    const int b  = blockIdx.y;
    const int q0 = blockIdx.x * 16;
    const int tid = threadIdx.x;
    const int h = tid >> 5, lane = tid & 31;
    const int lhi = lane >> 4, mcol = lane & 15;

    if (tid < 256) wx_s[tid] = Wx[tid];
    __syncthreads();
    float wrow[16];
#pragma unroll
    for (int hh = 0; hh < 16; ++hh) wrow[hh] = wx_s[h * 16 + hh];

    const __bf16* qh  = qk + (size_t)(b * 16 + h) * NCTX * 64;
    const __bf16* vhT = vT + (size_t)(b * 16 + h) * 64 * NCTX;

    v16bf aq0 = load_frag(qh + (size_t)(q0 + mcol) * 64, lhi);
    v16bf aq1 = load_frag(qh + (size_t)(q0 + mcol) * 64 + 32, lhi);

    float mrow[8], linv[8];
#pragma unroll
    for (int r = 0; r < 8; ++r) {
        int row = q0 + r + 8 * lhi;
        mrow[r] = mstat[(size_t)(b * 16 + h) * NCTX + row];
        linv[r] = 1.0f / lstat[(size_t)(b * 16 + h) * NCTX + row];
    }

    V8 o[4];
#pragma unroll
    for (int nt = 0; nt < 4; ++nt)
#pragma unroll
        for (int i = 0; i < 8; ++i) o[nt].e[i] = 0.f;

    for (int kc = 0; kc < NCTX; kc += 32) {
        if (kc + 32 < NCTX) {                            // prefetch next chunk
            __builtin_prefetch(qh + (size_t)(kc + 32 + mcol) * 64, 0, 3);
            __builtin_prefetch(qh + (size_t)(kc + 48 + mcol) * 64, 0, 3);
            __builtin_prefetch(vhT + (size_t)mcol * NCTX + kc + 32, 0, 3);
        }
        // S tiles for the two 16-col halves of this 32-wide k chunk
#pragma unroll
        for (int jt = 0; jt < 2; ++jt) {
            const int krow = kc + jt * 16 + mcol;
            v16bf b0 = load_frag(qh + (size_t)krow * 64, lhi);
            v16bf b1 = load_frag(qh + (size_t)krow * 64 + 32, lhi);
            V8 s;
#pragma unroll
            for (int i = 0; i < 8; ++i) s.e[i] = 0.f;
            s.v = wmma_bf16(aq0, b0, s.v);
            s.v = wmma_bf16(aq1, b1, s.v);
            const int jl = jt * 16 + mcol;
#pragma unroll
            for (int r = 0; r < 8; ++r) {
                float p = __expf(s.e[r] - mrow[r]) * linv[r];
                pbuf[(h * 16 + r + 8 * lhi) * PBS + jl] = p;
            }
        }
        __syncthreads();

        // Head-mix on VALU straight into the A fragment of Pmix @ V.
        // A halves are K-contiguous in runs of 4 -> float4 LDS reads.
        V16 pa;
#pragma unroll
        for (int g = 0; g < 4; ++g) {
            const int kb = ((g >> 1) << 4) + ((g & 1) << 2) + (lhi << 3);
            float a0 = 0.f, a1 = 0.f, a2 = 0.f, a3 = 0.f;
#pragma unroll
            for (int hh = 0; hh < 16; ++hh) {
                float4 pv = *(const float4*)&pbuf[(hh * 16 + mcol) * PBS + kb];
                a0 += wrow[hh] * pv.x;
                a1 += wrow[hh] * pv.y;
                a2 += wrow[hh] * pv.z;
                a3 += wrow[hh] * pv.w;
            }
            pa.e[4 * g + 0] = f2bf(a0);
            pa.e[4 * g + 1] = f2bf(a1);
            pa.e[4 * g + 2] = f2bf(a2);
            pa.e[4 * g + 3] = f2bf(a3);
        }
        // PV: B fragment from vT rows (d major) -> two b128 loads each
#pragma unroll
        for (int nt = 0; nt < 4; ++nt) {
            v16bf bv = load_frag(vhT + (size_t)(nt * 16 + mcol) * NCTX + kc, lhi);
            o[nt].v = wmma_bf16(pa.v, bv, o[nt].v);
        }
        __syncthreads();
    }

#pragma unroll
    for (int nt = 0; nt < 4; ++nt)
#pragma unroll
        for (int r = 0; r < 8; ++r) {
            int n = q0 + r + 8 * lhi;
            int d = h * 64 + nt * 16 + mcol;
            o1[((size_t)b * NCTX + n) * 1024 + d] = f2bf(o[nt].e[r]);
        }
}

// ---------------------------------------------------------------------------
extern "C" void kernel_launch(void* const* d_in, const int* in_sizes, int n_in,
                              void* d_out, int out_size, void* d_ws, size_t ws_size,
                              hipStream_t stream) {
    const int dims = 1024, head = 16, B = 2;
    const int M = B * NCTX;                              // 4096 rows

    const float* x    = (const float*)d_in[0];
    const float* ln_g = (const float*)d_in[1];
    const float* ln_b = (const float*)d_in[2];
    const float* W_kv = (const float*)d_in[3];
    const float* W_x  = (const float*)d_in[4];
    // d_in[5] = W_xa : dead code in reference (o2 discarded)
    const float* W_out = (const float*)d_in[6];

    char* ws = (char*)d_ws;
    size_t off = 0;
    __bf16* ws_h     = (__bf16*)(ws + off); off += (size_t)M * dims * 2;
    __bf16* ws_WkvT  = (__bf16*)(ws + off); off += (size_t)dims * 2 * dims * 2;
    __bf16* ws_WoutT = (__bf16*)(ws + off); off += (size_t)dims * dims * 2;
    __bf16* ws_qk    = (__bf16*)(ws + off); off += (size_t)B * head * NCTX * 64 * 2;
    __bf16* ws_vT    = (__bf16*)(ws + off); off += (size_t)B * head * NCTX * 64 * 2;
    __bf16* ws_o1    = (__bf16*)(ws + off); off += (size_t)M * dims * 2;
    float*  ws_m     = (float*)(ws + off);  off += (size_t)B * head * NCTX * 4;
    float*  ws_l     = (float*)(ws + off);  off += (size_t)B * head * NCTX * 4;
    (void)ws_size; (void)in_sizes; (void)n_in; (void)out_size;

    // one-time transposed weight conversions: W[K][N] -> WT[N][K] bf16
    k_cvt_t<<<dim3((2 * dims) / 256, dims), 256, 0, stream>>>(W_kv, ws_WkvT,
                                                              2 * dims, dims);
    k_cvt_t<<<dim3(dims / 256, dims), 256, 0, stream>>>(W_out, ws_WoutT,
                                                        dims, dims);

    k_ln<<<M, 256, 0, stream>>>(x, ln_g, ln_b, ws_h);

    // kv = h @ W_kv -> qk / vT
    k_gemm<<<dim3((2 * dims) / 64, M / 64), 256, 0, stream>>>(
        ws_h, ws_WkvT, M, 2 * dims, dims,
        nullptr, ws_qk, ws_vT, /*out_mode=*/2);

    k_stats<<<dim3(NCTX / 16, B * head), 256, 0, stream>>>(ws_qk, ws_m, ws_l);

    k_attn<<<dim3(NCTX / 16, B), 512, 0, stream>>>(
        ws_qk, ws_vT, ws_m, ws_l, W_x, ws_o1);

    // out = o1 @ W_out (f32)
    k_gemm<<<dim3(dims / 64, M / 64), 256, 0, stream>>>(
        ws_o1, ws_WoutT, M, dims, dims,
        (float*)d_out, nullptr, nullptr, /*out_mode=*/0);
}
